// GPT_45071386804490
// MI455X (gfx1250) — matrix operand order repaired and checked by reading, not compiled, not run on hardware
//
#include <hip/hip_runtime.h>
#include <hip/hip_bf16.h>
#include <cstdint>

#define L_   12
#define D_   1024
#define NH_  16
#define HD_  64
#define HID_ 2816
#define V_   50257
#define B_   8
#define T_   1024

typedef __bf16 bf16_t;
typedef __attribute__((ext_vector_type(16))) __bf16   v16bf;
typedef __attribute__((ext_vector_type(8)))  float    v8f;
typedef __attribute__((ext_vector_type(4)))  unsigned int u32x4;

union FragAB { v16bf v; u32x4 u[2]; };

__device__ __forceinline__ bf16_t f2bf(float f) {
  union { float f; unsigned u; } a; a.f = f;
  unsigned r = a.u + 0x7FFFu + ((a.u >> 16) & 1u);
  union { unsigned short s; bf16_t b; } o; o.s = (unsigned short)(r >> 16);
  return o.b;
}

// ---------------- Tensor Data Mover staging ----------------
#if __has_builtin(__builtin_amdgcn_tensor_load_to_lds)
#define USE_TDM 1
#else
#define USE_TDM 0
#endif

#if USE_TDM
typedef __attribute__((ext_vector_type(4))) unsigned int tdm_u32x4;
typedef __attribute__((ext_vector_type(8))) int          tdm_i32x8;
typedef __attribute__((ext_vector_type(4))) int          tdm_i32x4;

// DMA a 128x32 bf16 tile (row stride = rowStrideElems) from global into LDS,
// padded to a 40-element (80B) row pitch: pad_interval = 16 DWORDs (one 64B row),
// pad_amount = 4 DWORDs (16B). Tracked on TENSORcnt.
__device__ __forceinline__ void tdm_load_tile_128x32_bf16(
    unsigned lds_byte_off, const bf16_t* gsrc, int rowStrideElems) {
  unsigned long long ga = (unsigned long long)(size_t)gsrc;
  unsigned K = (unsigned)rowStrideElems;
  tdm_u32x4 g0;
  g0[0] = 1u;                                                  // count=1 (valid D#)
  g0[1] = lds_byte_off;                                        // lds_addr
  g0[2] = (unsigned)ga;                                        // global_addr[31:0]
  g0[3] = (unsigned)((ga >> 32) & 0x01FFFFFFu) | (2u << 30);   // addr[56:32] | type=2
  tdm_i32x8 g1;
  g1[0] = (int)((1u << 16)     // data_size = 2 bytes
              | (1u << 20)     // pad_enable
              | (3u << 22)     // pad_interval: 16 DWORDs
              | (3u << 25));   // pad_amount: 4 DWORDs  -> LDS pitch 40 elements
  g1[1] = (int)((K & 0xFFFFu) << 16);          // tensor_dim0[15:0]
  g1[2] = (int)((K >> 16) & 0xFFFFu);          // tensor_dim0[31:16] | tensor_dim1[15:0]=0
  g1[3] = (int)((1u << 4) | (32u << 16));      // tensor_dim1=1<<20 rows | tile_dim0=32
  g1[4] = (int)(128u);                         // tile_dim1=128 | tile_dim2=0
  g1[5] = (int)K;                              // tensor_dim0_stride[31:0]
  g1[6] = 0;                                   // stride0[47:32] | stride1[15:0]
  g1[7] = 0;
  tdm_i32x4 gz = {};
#if defined(__clang_major__) && (__clang_major__ >= 23)
  tdm_i32x8 gz8 = {};
  __builtin_amdgcn_tensor_load_to_lds(g0, g1, gz, gz, gz8, 0);
#else
  __builtin_amdgcn_tensor_load_to_lds(g0, g1, gz, gz, 0);
#endif
}
#endif  // USE_TDM

// ---------------- generic fp32 -> bf16 convert ----------------
__global__ void k_f32_to_bf16(const float* __restrict__ in, bf16_t* __restrict__ out, long n) {
  long i = (long)blockIdx.x * blockDim.x + threadIdx.x;
  if (i < n) out[i] = f2bf(in[i]);
}

// ---------------- embedding gather ----------------
__global__ void k_embed(const int* __restrict__ tok, const float* __restrict__ wte,
                        float* __restrict__ x) {
  long i = (long)blockIdx.x * blockDim.x + threadIdx.x;  // over B*T*D
  long row = i >> 10;
  int d = (int)(i & 1023);
  x[i] = wte[(long)tok[row] * D_ + d];
}

// ---------------- LayerNorm (row per block) ----------------
__global__ __launch_bounds__(256) void k_layernorm(
    const float* __restrict__ x, const float* __restrict__ w, const float* __restrict__ bia,
    bf16_t* __restrict__ obf, float* __restrict__ of32, int rowScale, int rowOff) {
  int row = blockIdx.x * rowScale + rowOff;
  const float4* xr = (const float4*)(x + (size_t)row * D_);
  float4 v = xr[threadIdx.x];
  float s = v.x + v.y + v.z + v.w;
  float q = v.x * v.x + v.y * v.y + v.z * v.z + v.w * v.w;
#pragma unroll
  for (int o = 16; o > 0; o >>= 1) { s += __shfl_xor(s, o, 32); q += __shfl_xor(q, o, 32); }
  __shared__ float rs[8], rq[8];
  int wv = threadIdx.x >> 5, ln = threadIdx.x & 31;
  if (ln == 0) { rs[wv] = s; rq[wv] = q; }
  __syncthreads();
  float ts = 0.f, tq2 = 0.f;
#pragma unroll
  for (int i = 0; i < 8; ++i) { ts += rs[i]; tq2 += rq[i]; }
  float mean = ts * (1.0f / D_);
  float var = tq2 * (1.0f / D_) - mean * mean;
  float inv = rsqrtf(var + 1e-5f);
  float4 wv4 = ((const float4*)w)[threadIdx.x];
  float4 bv4 = ((const float4*)bia)[threadIdx.x];
  float o0 = (v.x - mean) * inv * wv4.x + bv4.x;
  float o1 = (v.y - mean) * inv * wv4.y + bv4.y;
  float o2 = (v.z - mean) * inv * wv4.z + bv4.z;
  float o3 = (v.w - mean) * inv * wv4.w + bv4.w;
  size_t ob = (size_t)blockIdx.x * D_ + (size_t)threadIdx.x * 4;
  if (obf) { obf[ob] = f2bf(o0); obf[ob + 1] = f2bf(o1); obf[ob + 2] = f2bf(o2); obf[ob + 3] = f2bf(o3); }
  if (of32) { of32[ob] = o0; of32[ob + 1] = o1; of32[ob + 2] = o2; of32[ob + 3] = o3; }
}

// ---------------- WMMA bf16 GEMM with TDM double-buffered staging ----------------
// out[M,N](fp32) = A[M,K](bf16) * Wt[N,K](bf16)^T (+ bias[N]) (+ res[M,N])
// Block: 256 thr (8 waves), tile 128x128, k-step 32. Wave = 32x64 patch (2x4 accs).
// All dims are multiples of 128 (M,N) / 32 (K) for this model.
__global__ __launch_bounds__(256) void k_gemm_bf16(
    const bf16_t* __restrict__ A, const bf16_t* __restrict__ Wt,
    const float* __restrict__ bias, const float* __restrict__ res,
    float* __restrict__ out, int M, int N, int K) {
  __shared__ __align__(16) bf16_t shA[2][128 * 40];  // 40-elem pitch: conflict-free b128
  __shared__ __align__(16) bf16_t shB[2][128 * 40];
  const int t = threadIdx.x;
  const int lane = t & 31, wv = t >> 5;
  const int wr = (wv >> 1) * 32;   // wave row base in tile
  const int wc = (wv & 1) * 64;    // wave col base in tile
  const int lr = lane & 15;
  const int lh = lane >> 4;
  const long blockM = (long)blockIdx.y * 128;
  const long blockN = (long)blockIdx.x * 128;

  v8f acc[2][4] = {};
  const int nIter = K >> 5;

#if USE_TDM
  const unsigned ldsA0 = (unsigned)(size_t)(&shA[0][0]);
  const unsigned ldsB0 = (unsigned)(size_t)(&shB[0][0]);
  const bf16_t* Abase = A + blockM * (long)K;
  const bf16_t* Bbase = Wt + blockN * (long)K;
  if (wv == 0) {
    tdm_load_tile_128x32_bf16(ldsA0, Abase, K);
    tdm_load_tile_128x32_bf16(ldsB0, Bbase, K);
  }
#endif

  for (int it = 0; it < nIter; ++it) {
    const int cur = it & 1;
#if USE_TDM
    if (wv == 0) {
      if (it + 1 < nIter) {  // DMA next stage while computing this one
        const unsigned soff = (unsigned)((cur ^ 1) * 128 * 40 * 2);
        tdm_load_tile_128x32_bf16(ldsA0 + soff, Abase + (long)(it + 1) * 32, K);
        tdm_load_tile_128x32_bf16(ldsB0 + soff, Bbase + (long)(it + 1) * 32, K);
        __builtin_amdgcn_s_wait_tensorcnt(2);  // current stage's 2 ops retired
      } else {
        __builtin_amdgcn_s_wait_tensorcnt(0);
      }
    }
    __syncthreads();
    const bf16_t* sA = shA[cur];
    const bf16_t* sB = shB[cur];
#else
    const int k0 = it * 32;
#pragma unroll
    for (int i = 0; i < 2; ++i) {
      int c = t + i * 256;
      int row = c >> 2, ch = c & 3;
      const u32x4* ga = (const u32x4*)(A + (blockM + row) * (long)K + k0 + ch * 8);
      *(u32x4*)(&shA[0][0] + row * 40 + ch * 8) = *ga;
      const u32x4* gb = (const u32x4*)(Wt + (blockN + row) * (long)K + k0 + ch * 8);
      *(u32x4*)(&shB[0][0] + row * 40 + ch * 8) = *gb;
    }
    __syncthreads();
    const bf16_t* sA = &shA[0][0];
    const bf16_t* sB = &shB[0][0];
    (void)cur;
#endif
    FragAB af[2], bfm[4];
#pragma unroll
    for (int mr = 0; mr < 2; ++mr) {
      // A 16x32 bf16 layout: lanes 0-15 -> K[0..7]&[16..23], lanes 16-31 -> K[8..15]&[24..31]
      const bf16_t* p = sA + (wr + mr * 16 + lr) * 40 + lh * 8;
      af[mr].u[0] = *(const u32x4*)(p);
      af[mr].u[1] = *(const u32x4*)(p + 16);
    }
#pragma unroll
    for (int nc = 0; nc < 4; ++nc) {
      // B 32x16 bf16 layout: lanes 0-15 -> K[0..7]&[16..23], lanes 16-31 -> K[8..15]&[24..31] of col lr
      const bf16_t* p = sB + (wc + nc * 16 + lr) * 40 + lh * 8;
      bfm[nc].u[0] = *(const u32x4*)(p);
      bfm[nc].u[1] = *(const u32x4*)(p + 16);
    }
#pragma unroll
    for (int mr = 0; mr < 2; ++mr)
#pragma unroll
      for (int nc = 0; nc < 4; ++nc)
        acc[mr][nc] = __builtin_amdgcn_wmma_f32_16x16x32_bf16(
            false, af[mr].v, false, bfm[nc].v, (short)0, acc[mr][nc], false, false);
    __syncthreads();  // all waves done reading this stage before its buffer is re-DMA'd
  }

  // epilogue: C/D layout -> lane l, vgpr r : row = r + 8*(l>>4), col = l&15
#pragma unroll
  for (int mr = 0; mr < 2; ++mr) {
#pragma unroll
    for (int nc = 0; nc < 4; ++nc) {
      long col = blockN + wc + nc * 16 + lr;
      float bv = bias ? bias[col] : 0.0f;
#pragma unroll
      for (int r = 0; r < 8; ++r) {
        long row = blockM + wr + mr * 16 + r + 8 * lh;
        long oi = row * (long)N + col;
        float vv = acc[mr][nc][r] + bv;
        if (res) vv += res[oi];
        out[oi] = vv;
      }
    }
  }
}

// ---------------- RoPE on q,k inside qkv buffer ----------------
__global__ void k_rope(float* __restrict__ qkv) {
  long i = (long)blockIdx.x * blockDim.x + threadIdx.x;  // over B*T*NH*32
  int d = (int)(i & 31);
  int h = (int)((i >> 5) & (NH_ - 1));
  int tt = (int)((i >> 9) & (T_ - 1));
  int b = (int)(i >> 19);
  float invf = __expf(-(float)d * 0.2878231366242557f);  // 10000^(-d/32)
  float ang = (float)tt * invf;
  float s, c;
  __sincosf(ang, &s, &c);
  size_t base = ((size_t)(b * T_ + tt)) * (3 * D_) + (size_t)h * HD_ + d;
  float q1 = qkv[base], q2 = qkv[base + 32];
  qkv[base] = q1 * c - q2 * s;
  qkv[base + 32] = q2 * c + q1 * s;
  float k1 = qkv[base + D_], k2 = qkv[base + D_ + 32];
  qkv[base + D_] = k1 * c - k2 * s;
  qkv[base + D_ + 32] = k2 * c + k1 * s;
}

// ---------------- attention: wave per query row, online softmax ----------------
__global__ __launch_bounds__(256) void k_attn(const float* __restrict__ qkv,
                                              bf16_t* __restrict__ out) {
  __shared__ __align__(16) float qs[8][HD_];
  int wv = threadIdx.x >> 5, lane = threadIdx.x & 31;
  int gid = blockIdx.x * 8 + wv;
  int tq = gid & (T_ - 1);
  int h = (gid >> 10) & (NH_ - 1);
  int b = gid >> 14;
  const size_t rs3 = 3 * D_;
  const float* qrow = qkv + ((size_t)(b * T_ + tq)) * rs3 + (size_t)h * HD_;
  qs[wv][lane] = qrow[lane];
  qs[wv][lane + 32] = qrow[lane + 32];
  __syncthreads();
  float m = -3.0e38f, ssum = 0.0f, acc0 = 0.0f, acc1 = 0.0f;
  const float* kbase = qkv + D_ + h * HD_;
  const float* vbase = qkv + 2 * D_ + h * HD_;
  for (int kb = 0; kb <= tq; kb += 32) {
    int tk = kb + lane;
    float sc = -3.0e38f;
    if (tk <= tq) {
      const float4* kr = (const float4*)(kbase + (size_t)(b * T_ + tk) * rs3);
      const float4* qq = (const float4*)qs[wv];
      float s = 0.0f;
#pragma unroll
      for (int d4 = 0; d4 < 16; ++d4) {
        float4 kv = kr[d4];
        float4 qv = qq[d4];
        s += qv.x * kv.x + qv.y * kv.y + qv.z * kv.z + qv.w * kv.w;
      }
      sc = s * 0.125f;  // 1/sqrt(64)
    }
    float cm = sc;
#pragma unroll
    for (int o = 16; o > 0; o >>= 1) cm = fmaxf(cm, __shfl_xor(cm, o, 32));
    float nm = fmaxf(m, cm);
    float p = (tk <= tq) ? __expf(sc - nm) : 0.0f;
    float ps = p;
#pragma unroll
    for (int o = 16; o > 0; o >>= 1) ps += __shfl_xor(ps, o, 32);
    float corr = __expf(m - nm);
    ssum = ssum * corr + ps;
    acc0 *= corr;
    acc1 *= corr;
    int jmax = min(32, tq - kb + 1);
    for (int j = 0; j < jmax; ++j) {
      float pj = __shfl(p, j, 32);
      const float* vr = vbase + (size_t)(b * T_ + kb + j) * rs3;
      acc0 += pj * vr[lane];
      acc1 += pj * vr[lane + 32];
    }
    m = nm;
  }
  float inv = 1.0f / ssum;
  size_t ob = ((size_t)(b * T_ + tq)) * D_ + (size_t)h * HD_;
  out[ob + lane] = f2bf(acc0 * inv);
  out[ob + lane + 32] = f2bf(acc1 * inv);
}

// ---------------- SwiGLU elementwise ----------------
__global__ void k_swiglu(const float* __restrict__ u, const float* __restrict__ g,
                         bf16_t* __restrict__ o, long n) {
  long i = (long)blockIdx.x * blockDim.x + threadIdx.x;
  if (i < n) {
    float x = u[i];
    float s = x / (1.0f + __expf(-x));
    o[i] = f2bf(s * g[i]);
  }
}

// ---------------- last-token logits: wave per vocab row ----------------
__global__ __launch_bounds__(256) void k_logits(const float* __restrict__ xl,
                                                const float* __restrict__ wte,
                                                float* __restrict__ out) {
  int wv = threadIdx.x >> 5, lane = threadIdx.x & 31;
  int v = blockIdx.x * 8 + wv;
  if (v >= V_) return;
  float acc[B_] = {};
  const float* wr = wte + (size_t)v * D_;
  for (int c = 0; c < 32; ++c) {
    float wvv = wr[c * 32 + lane];
#pragma unroll
    for (int b = 0; b < B_; ++b) acc[b] += wvv * xl[b * D_ + c * 32 + lane];
  }
#pragma unroll
  for (int b = 0; b < B_; ++b) {
    float s = acc[b];
#pragma unroll
    for (int o = 16; o > 0; o >>= 1) s += __shfl_xor(s, o, 32);
    if (lane == 0) out[(size_t)b * V_ + v] = s;
  }
}

extern "C" void kernel_launch(void* const* d_in, const int* in_sizes, int n_in,
                              void* d_out, int out_size, void* d_ws, size_t ws_size,
                              hipStream_t stream) {
  const int* tokens     = (const int*)d_in[0];
  const float* wte      = (const float*)d_in[1];
  const float* c_attn_w = (const float*)d_in[2];
  const float* c_attn_b = (const float*)d_in[3];
  const float* c_proj_w = (const float*)d_in[4];
  const float* c_proj_b = (const float*)d_in[5];
  const float* ln1_w    = (const float*)d_in[6];
  const float* ln1_b    = (const float*)d_in[7];
  const float* ln2_w    = (const float*)d_in[8];
  const float* ln2_b    = (const float*)d_in[9];
  const float* w1       = (const float*)d_in[10];
  const float* w3       = (const float*)d_in[11];
  const float* w2       = (const float*)d_in[12];
  const float* lnf_w    = (const float*)d_in[13];
  const float* lnf_b    = (const float*)d_in[14];
  float* logits = (float*)d_out;

  char* ws = (char*)d_ws;
  size_t off = 0;
  auto take = [&](size_t bytes) -> char* {
    off = (off + 255) & ~(size_t)255;
    char* p = ws + off;
    off += bytes;
    return p;
  };
  const size_t NTOK = (size_t)B_ * T_;
  bf16_t* caw_bf = (bf16_t*)take(sizeof(bf16_t) * (size_t)L_ * 3 * D_ * D_);
  bf16_t* cpw_bf = (bf16_t*)take(sizeof(bf16_t) * (size_t)L_ * D_ * D_);
  bf16_t* w1_bf  = (bf16_t*)take(sizeof(bf16_t) * (size_t)L_ * HID_ * D_);
  bf16_t* w3_bf  = (bf16_t*)take(sizeof(bf16_t) * (size_t)L_ * HID_ * D_);
  bf16_t* w2_bf  = (bf16_t*)take(sizeof(bf16_t) * (size_t)L_ * D_ * HID_);
  float* x       = (float*)take(sizeof(float) * NTOK * D_);
  bf16_t* hbf    = (bf16_t*)take(sizeof(bf16_t) * NTOK * D_);
  float* qkv     = (float*)take(sizeof(float) * NTOK * 3 * D_);
  bf16_t* attnb  = (bf16_t*)take(sizeof(bf16_t) * NTOK * D_);
  float* ubuf    = (float*)take(sizeof(float) * NTOK * HID_);
  float* gbuf    = (float*)take(sizeof(float) * NTOK * HID_);
  bf16_t* mbf    = (bf16_t*)take(sizeof(bf16_t) * NTOK * HID_);
  float* xlast   = (float*)take(sizeof(float) * B_ * D_);

  auto conv = [&](const float* src, bf16_t* dst, long n) {
    k_f32_to_bf16<<<(unsigned)((n + 255) / 256), 256, 0, stream>>>(src, dst, n);
  };
  conv(c_attn_w, caw_bf, (long)L_ * 3 * D_ * D_);
  conv(c_proj_w, cpw_bf, (long)L_ * D_ * D_);
  conv(w1, w1_bf, (long)L_ * HID_ * D_);
  conv(w3, w3_bf, (long)L_ * HID_ * D_);
  conv(w2, w2_bf, (long)L_ * D_ * HID_);

  k_embed<<<(unsigned)(NTOK * D_ / 256), 256, 0, stream>>>(tokens, wte, x);

  for (int l = 0; l < L_; ++l) {
    k_layernorm<<<(unsigned)NTOK, 256, 0, stream>>>(x, ln1_w + l * D_, ln1_b + l * D_,
                                                    hbf, (float*)nullptr, 1, 0);
    {
      dim3 g(3 * D_ / 128, (unsigned)(NTOK / 128));
      k_gemm_bf16<<<g, 256, 0, stream>>>(hbf, caw_bf + (size_t)l * 3 * D_ * D_,
                                         c_attn_b + (size_t)l * 3 * D_, nullptr, qkv,
                                         (int)NTOK, 3 * D_, D_);
    }
    k_rope<<<(unsigned)(NTOK * NH_ * 32 / 256), 256, 0, stream>>>(qkv);
    k_attn<<<(unsigned)((size_t)B_ * NH_ * T_ / 8), 256, 0, stream>>>(qkv, attnb);
    {
      dim3 g(D_ / 128, (unsigned)(NTOK / 128));
      k_gemm_bf16<<<g, 256, 0, stream>>>(attnb, cpw_bf + (size_t)l * D_ * D_,
                                         c_proj_b + (size_t)l * D_, x, x,
                                         (int)NTOK, D_, D_);
    }
    k_layernorm<<<(unsigned)NTOK, 256, 0, stream>>>(x, ln2_w + l * D_, ln2_b + l * D_,
                                                    hbf, (float*)nullptr, 1, 0);
    {
      dim3 g(HID_ / 128, (unsigned)(NTOK / 128));
      k_gemm_bf16<<<g, 256, 0, stream>>>(hbf, w1_bf + (size_t)l * HID_ * D_, nullptr,
                                         nullptr, ubuf, (int)NTOK, HID_, D_);
      k_gemm_bf16<<<g, 256, 0, stream>>>(hbf, w3_bf + (size_t)l * HID_ * D_, nullptr,
                                         nullptr, gbuf, (int)NTOK, HID_, D_);
    }
    k_swiglu<<<(unsigned)(NTOK * HID_ / 256), 256, 0, stream>>>(ubuf, gbuf, mbf,
                                                                (long)NTOK * HID_);
    {
      dim3 g(D_ / 128, (unsigned)(NTOK / 128));
      k_gemm_bf16<<<g, 256, 0, stream>>>(mbf, w2_bf + (size_t)l * D_ * HID_, nullptr, x,
                                         x, (int)NTOK, D_, HID_);
    }
  }
  k_layernorm<<<B_, 256, 0, stream>>>(x, lnf_w, lnf_b, (bf16_t*)nullptr, xlast, T_, T_ - 1);
  k_logits<<<(V_ + 7) / 8, 256, 0, stream>>>(xlast, wte, logits);
}
